// HardSimMask_27762668601763
// MI455X (gfx1250) — matrix operand order repaired and checked by reading, not compiled, not run on hardware
//
#include <hip/hip_runtime.h>
#include <hip/hip_bf16.h>

// ---------------------------------------------------------------------------
// HardSimMask on MI455X (gfx1250): masked -L2-cdist as bf16 WMMA GEMM + epilogue
//   sim = -sqrt(max(x2_i + y2_j - 2 * X @ Xn^T, 0)); out = sim < -8 ? -inf : sim
// N = M = 8192, D = 256.
// Round 4: fix NT-load operand type (clang ext_vector float4, not HIP struct).
// Async global->LDS staging in GVS form (SGPR base + 32-bit VGPR offset);
// non-temporal epilogue stores keep L2 for the reusable bf16 operands.
// ---------------------------------------------------------------------------

typedef __attribute__((ext_vector_type(16))) __bf16 v16bf;
typedef __attribute__((ext_vector_type(8)))  __bf16 v8bf;
typedef __attribute__((ext_vector_type(8)))  float  v8f;
typedef __attribute__((ext_vector_type(4)))  float  v4f;

#define NROWS 8192
#define DIM   256
#define MIN_SIM (-8.0f)   // (0 - (-sqrt(256))) * 0.5 + (-sqrt(256))

__device__ __forceinline__ unsigned int pack_bf16x2(float a, float b) {
    unsigned int ua = __float_as_uint(a);
    unsigned int ub = __float_as_uint(b);
    ua = (ua + 0x7FFFu + ((ua >> 16) & 1u)) >> 16;   // RNE f32 -> bf16
    ub = (ub + 0x7FFFu + ((ub >> 16) & 1u)) >> 16;
    return ua | (ub << 16);
}

// One wave per row: sum of squares (f32) + packed bf16 conversion.
__global__ void __launch_bounds__(256)
hsm_prep_kernel(const float* __restrict__ in,
                float* __restrict__ norms,
                unsigned short* __restrict__ bf) {
    const int lane = threadIdx.x & 31;
    const int wave = threadIdx.x >> 5;
    const int row  = blockIdx.x * 8 + wave;

    const v4f* p = (const v4f*)(in + (size_t)row * DIM + lane * 8);
    v4f a = __builtin_nontemporal_load(p);       // one-shot reads: NT
    v4f b = __builtin_nontemporal_load(p + 1);

    float s = a.x * a.x + a.y * a.y + a.z * a.z + a.w * a.w
            + b.x * b.x + b.y * b.y + b.z * b.z + b.w * b.w;

    uint4 pk;
    pk.x = pack_bf16x2(a.x, a.y);
    pk.y = pack_bf16x2(a.z, a.w);
    pk.z = pack_bf16x2(b.x, b.y);
    pk.w = pack_bf16x2(b.z, b.w);
    *(uint4*)(bf + (size_t)row * DIM + lane * 8) = pk;   // reused: keep RT

    #pragma unroll
    for (int off = 16; off > 0; off >>= 1)
        s += __shfl_xor(s, off, 32);
    if (lane == 0) norms[row] = s;
}

// Direct global->LDS async 16B copy, GVS addressing: SGPR64 base + VGPR32
// byte offset. LDS operand is the wave-relative byte offset (= low 32 bits
// of the generic pointer; generic->LDS addrspacecast is a truncate).
__device__ __forceinline__ void async_copy_b128(unsigned int lds_off,
                                                unsigned int voff,
                                                const unsigned short* base) {
    asm volatile("global_load_async_to_lds_b128 %0, %1, %2"
                 :: "v"(lds_off), "v"(voff), "s"(base)
                 : "memory");
}

__device__ __forceinline__ void async_wait0() {
    asm volatile("s_wait_asynccnt 0x0" ::: "memory");
}

// ---------------------------------------------------------------------------
// GEMM: 128x128 block tile, 8 waves (2x4), each wave 4x2 tiles of 16x16.
// K staged through double-buffered LDS in chunks of 32 bf16 via async copies.
// ---------------------------------------------------------------------------
#define BK         32
#define LDS_STRIDE 40   // 32 + 8 pad: keeps 16B alignment, spreads LDS banks
#define KSTEPS     (DIM / BK)

__global__ void __launch_bounds__(256)
hsm_gemm_kernel(const unsigned short* __restrict__ xbf,
                const unsigned short* __restrict__ xnbf,
                const float* __restrict__ x2,
                const float* __restrict__ y2,
                float* __restrict__ out) {
    __shared__ unsigned short As[2][128 * LDS_STRIDE];
    __shared__ unsigned short Bs[2][128 * LDS_STRIDE];

    const int tid  = threadIdx.x;
    const int lane = tid & 31;
    const int wave = tid >> 5;
    const int wm   = wave >> 2;          // 0..1
    const int wn   = wave & 3;           // 0..3
    const int l16  = lane & 15;
    const int lhi  = lane >> 4;          // 0 or 1

    const int rowBase = blockIdx.y * 128;
    const int colBase = blockIdx.x * 128;

    v8f acc[4][2];
    const v8f vzero = {0.f, 0.f, 0.f, 0.f, 0.f, 0.f, 0.f, 0.f};
    #pragma unroll
    for (int i = 0; i < 4; ++i)
        #pragma unroll
        for (int j = 0; j < 2; ++j)
            acc[i][j] = vzero;

    // 128 rows x 32 bf16 per tile = 512 16B-chunks; 256 threads x 2 chunks
    // each, for both A (x rows) and B (xn rows). Precompute per-thread
    // 32-bit global byte offsets (k0 advance is +64B) and LDS destinations.
    unsigned int gvA[2], gvB[2];             // global byte offsets at k0 = 0
    unsigned int ldsA[2][2], ldsB[2][2];     // [buf][chunk] LDS byte offsets
    #pragma unroll
    for (int i = 0; i < 2; ++i) {
        const int chunk = tid + i * 256;     // 0..511
        const int r = chunk >> 2;
        const int c = (chunk & 3) * 8;
        gvA[i] = ((unsigned)(rowBase + r) * DIM + c) * 2u;
        gvB[i] = ((unsigned)(colBase + r) * DIM + c) * 2u;
        #pragma unroll
        for (int b = 0; b < 2; ++b) {
            ldsA[b][i] = (unsigned int)(uintptr_t)&As[b][r * LDS_STRIDE + c];
            ldsB[b][i] = (unsigned int)(uintptr_t)&Bs[b][r * LDS_STRIDE + c];
        }
    }

    auto issue_tile_copy = [&](int buf, int kbytes) {
        #pragma unroll
        for (int i = 0; i < 2; ++i) {
            async_copy_b128(ldsA[buf][i], gvA[i] + kbytes, xbf);
            async_copy_b128(ldsB[buf][i], gvB[i] + kbytes, xnbf);
        }
    };

    issue_tile_copy(0, 0);   // prologue: prefetch first K-tile

    #pragma unroll
    for (int kk = 0; kk < KSTEPS; ++kk) {
        const int buf = kk & 1;
        async_wait0();        // this wave's copies into `buf` have landed
        __syncthreads();      // every wave's copies have landed
        if (kk + 1 < KSTEPS)  // overlap next tile's copy with this compute
            issue_tile_copy(buf ^ 1, (kk + 1) * BK * 2);

        // A fragment layout (16-bit 16x32): lanes 0-15 hold K {0..7,16..23},
        // lanes 16-31 hold K {8..15,24..31}, M = lane%16.
        const int ka = lhi * 8;
        v16bf afrag[4];
        #pragma unroll
        for (int tm = 0; tm < 4; ++tm) {
            const int m = wm * 64 + tm * 16 + l16;
            v8bf lo = *(const v8bf*)&As[buf][m * LDS_STRIDE + ka];
            v8bf hi = *(const v8bf*)&As[buf][m * LDS_STRIDE + ka + 16];
            afrag[tm] = __builtin_shufflevector(lo, hi,
                0, 1, 2, 3, 4, 5, 6, 7, 8, 9, 10, 11, 12, 13, 14, 15);
        }

        // B fragment (32x16, B = Xn^T): N = lane%16, lanes 0-15 hold K {0..7,16..23},
        // lanes 16-31 hold K {8..15,24..31} — same K order as A.
        const int kb = lhi * 8;
        v16bf bfrag[2];
        #pragma unroll
        for (int tn = 0; tn < 2; ++tn) {
            const int n = wn * 32 + tn * 16 + l16;
            v8bf lo = *(const v8bf*)&Bs[buf][n * LDS_STRIDE + kb];
            v8bf hi = *(const v8bf*)&Bs[buf][n * LDS_STRIDE + kb + 16];
            bfrag[tn] = __builtin_shufflevector(lo, hi,
                0, 1, 2, 3, 4, 5, 6, 7, 8, 9, 10, 11, 12, 13, 14, 15);
        }

        #pragma unroll
        for (int tm = 0; tm < 4; ++tm)
            #pragma unroll
            for (int tn = 0; tn < 2; ++tn)
                acc[tm][tn] = __builtin_amdgcn_wmma_f32_16x16x32_bf16(
                    false, afrag[tm], false, bfrag[tn],
                    (short)0, acc[tm][tn], false, false);
    }

    // Epilogue: d2 = x2 + y2 - 2*dot; sim = -sqrt(max(d2,0)); mask < -8 -> -inf
    // C/D layout: col = lane%16, row = vgpr + 8*(lane>=16).
    // Output is write-once: non-temporal stores keep L2 for the bf16 operands.
    #pragma unroll
    for (int tm = 0; tm < 4; ++tm) {
        #pragma unroll
        for (int tn = 0; tn < 2; ++tn) {
            const int col = colBase + wn * 32 + tn * 16 + l16;
            const float yn = y2[col];
            #pragma unroll
            for (int v = 0; v < 8; ++v) {
                const int row = rowBase + wm * 64 + tm * 16 + v + 8 * lhi;
                float d2 = x2[row] + yn - 2.0f * acc[tm][tn][v];
                d2 = fmaxf(d2, 0.0f);
                float sim = -__builtin_sqrtf(d2);
                float res = (sim < MIN_SIM) ? -__builtin_inff() : sim;
                __builtin_nontemporal_store(res, &out[(size_t)row * NROWS + col]);
            }
        }
    }
}

extern "C" void kernel_launch(void* const* d_in, const int* in_sizes, int n_in,
                              void* d_out, int out_size, void* d_ws, size_t ws_size,
                              hipStream_t stream) {
    (void)in_sizes; (void)n_in; (void)out_size; (void)ws_size;

    const float* x   = (const float*)d_in[0];
    const float* x_n = (const float*)d_in[1];
    float* out = (float*)d_out;

    // Workspace layout: [x2: 8192 f32][y2: 8192 f32][pad to 64KB]
    //                   [xbf: 8192*256 bf16][xnbf: 8192*256 bf16]  (~8.1 MB)
    float* x2 = (float*)d_ws;
    float* y2 = x2 + NROWS;
    unsigned short* xbf  = (unsigned short*)((char*)d_ws + 65536);
    unsigned short* xnbf = xbf + (size_t)NROWS * DIM;

    hsm_prep_kernel<<<NROWS / 8, 256, 0, stream>>>(x, x2, xbf);
    hsm_prep_kernel<<<NROWS / 8, 256, 0, stream>>>(x_n, y2, xnbf);

    dim3 grid(NROWS / 128, NROWS / 128);
    hsm_gemm_kernel<<<grid, 256, 0, stream>>>(xbf, xnbf, x2, y2, out);
}
